// DiffAttn_72490458022170
// MI455X (gfx1250) — matrix operand order repaired and checked by reading, not compiled, not run on hardware
//
#include <hip/hip_runtime.h>
#include <hip/hip_bf16.h>
#include <stdint.h>

// Differential attention for MI455X (gfx1250), wave32 + WMMA bf16 + async-LDS.
//   B=2, T=4096, C=2048, H=128, D=64, DEPTH=2
// Pipeline:
//   1) qkv_proj_fused: q/k/v = x @ W^T via v_wmma_f32_16x16x32_bf16; x tile
//      loaded once for all three outputs. q,k stored [tok][h] bf16; v stored
//      TRANSPOSED [b][h][t] bf16 so attention V tiles are straight copies.
//   2) lam_kernel: lambda scalar.
//   3) diffattn_kernel: fused flash attention, two online softmaxes, two WMMA
//      output accumulators; K/V^T tiles fetched with
//      global_load_async_to_lds_b128 (ASYNCcnt). out = O1/l1 - lam*O2/l2.

typedef __bf16 bf16_t;
typedef __attribute__((ext_vector_type(16))) __bf16 v16bf;
typedef __attribute__((ext_vector_type(8)))  __bf16 v8bf;
typedef __attribute__((ext_vector_type(8)))  float  v8f;

__device__ __forceinline__ v8f wmma_bf16(v16bf a, v16bf b, v8f c) {
  return __builtin_amdgcn_wmma_f32_16x16x32_bf16(false, a, false, b, (short)0, c,
                                                 false, false);
}

// A-matrix 16x32 bf16 fragment (ISA 7.12.2): lane L holds row M=L&15,
// K(e) = (e>>3)*16 + (L>>4)*8 + (e&7). Source row-major [M][K], K contiguous.
__device__ __forceinline__ v16bf load_a_frag(const bf16_t* base, int stride,
                                             int kb, int lane) {
  const int row = lane & 15, half = lane >> 4;
  const bf16_t* p0 = base + row * stride + kb + half * 8;
  v8bf lo = *(const v8bf*)p0;
  v8bf hi = *(const v8bf*)(p0 + 16);
  v16bf r;
#pragma unroll
  for (int i = 0; i < 8; ++i) { r[i] = lo[i]; r[i + 8] = hi[i]; }
  return r;
}

// B-matrix 32x16 bf16 fragment: lane L holds column N=L&15, K(e)=(e>>3)*16+(L>>4)*8+(e&7).
// Source stores B^T: [N][K], K contiguous.
__device__ __forceinline__ v16bf load_bt_frag(const bf16_t* base, int stride,
                                              int kb, int lane) {
  const int col = lane & 15, half = lane >> 4;
  const bf16_t* p = base + col * stride + kb + half * 8;
  v8bf lo = *(const v8bf*)p;
  v8bf hi = *(const v8bf*)(p + 16);
  v16bf r;
#pragma unroll
  for (int i = 0; i < 8; ++i) { r[i] = lo[i]; r[i + 8] = hi[i]; }
  return r;
}

// Async global->LDS copy of 16 bytes per lane (CDNA5, tracked by ASYNCcnt).
__device__ __forceinline__ void async_copy_b128(const bf16_t* gaddr,
                                                const bf16_t* lds_ptr) {
  uint32_t l = (uint32_t)(uintptr_t)lds_ptr;  // low 32 bits = LDS offset
  asm volatile("global_load_async_to_lds_b128 %0, %1, off"
               :: "v"(l), "v"(gaddr) : "memory");
}
__device__ __forceinline__ void wait_async0() {
  asm volatile("s_wait_asynccnt 0x0" ::: "memory");
}

// ---------------- fused projection ------------------------------------------
// q[m][n] = sum_k x[m][k]*Wq[n][k]; same for k,v. Block: 128(M) x 128(N),
// K-step 32, 8 waves, wave = 16 rows x 128 cols x {q,k,v}.
// v is stored transposed: vo[b][h][t].
__global__ __launch_bounds__(256) void qkv_proj_fused(
    const float* __restrict__ x, const float* __restrict__ Wq,
    const float* __restrict__ Wk, const float* __restrict__ Wv,
    bf16_t* __restrict__ qo, bf16_t* __restrict__ ko, bf16_t* __restrict__ vo,
    int K, int T) {
  __shared__ __align__(16) bf16_t sX[128][32];
  __shared__ __align__(16) bf16_t sWq[128][32];
  __shared__ __align__(16) bf16_t sWk[128][32];
  __shared__ __align__(16) bf16_t sWv[128][32];
  const int tid = threadIdx.x, lane = tid & 31, wave = tid >> 5;
  const int half = lane >> 4, coln = lane & 15;
  const int m0 = blockIdx.x * 128;

  v8f accq[8], acck[8], accv[8];
#pragma unroll
  for (int j = 0; j < 8; ++j)
#pragma unroll
    for (int r = 0; r < 8; ++r) { accq[j][r] = 0.f; acck[j][r] = 0.f; accv[j][r] = 0.f; }

  for (int k0 = 0; k0 < K; k0 += 32) {
    float4 rx[4], rq[4], rk[4], rv[4];
#pragma unroll
    for (int i = 0; i < 4; ++i) {  // issue all 16 loads, then convert
      int pos = tid + 256 * i;
      int mm = pos >> 3, kk = (pos & 7) << 2;
      rx[i] = *(const float4*)(x  + (size_t)(m0 + mm) * K + k0 + kk);
      rq[i] = *(const float4*)(Wq + (size_t)mm * K + k0 + kk);
      rk[i] = *(const float4*)(Wk + (size_t)mm * K + k0 + kk);
      rv[i] = *(const float4*)(Wv + (size_t)mm * K + k0 + kk);
    }
#pragma unroll
    for (int i = 0; i < 4; ++i) {
      int pos = tid + 256 * i;
      int mm = pos >> 3, kk = (pos & 7) << 2;
      sX[mm][kk+0]=(bf16_t)rx[i].x; sX[mm][kk+1]=(bf16_t)rx[i].y;
      sX[mm][kk+2]=(bf16_t)rx[i].z; sX[mm][kk+3]=(bf16_t)rx[i].w;
      sWq[mm][kk+0]=(bf16_t)rq[i].x; sWq[mm][kk+1]=(bf16_t)rq[i].y;
      sWq[mm][kk+2]=(bf16_t)rq[i].z; sWq[mm][kk+3]=(bf16_t)rq[i].w;
      sWk[mm][kk+0]=(bf16_t)rk[i].x; sWk[mm][kk+1]=(bf16_t)rk[i].y;
      sWk[mm][kk+2]=(bf16_t)rk[i].z; sWk[mm][kk+3]=(bf16_t)rk[i].w;
      sWv[mm][kk+0]=(bf16_t)rv[i].x; sWv[mm][kk+1]=(bf16_t)rv[i].y;
      sWv[mm][kk+2]=(bf16_t)rv[i].z; sWv[mm][kk+3]=(bf16_t)rv[i].w;
    }
    __syncthreads();
    v16bf af = load_a_frag(&sX[wave * 16][0], 32, 0, lane);
#pragma unroll
    for (int j = 0; j < 8; ++j) {
      v16bf bq = load_bt_frag(&sWq[j * 16][0], 32, 0, lane);
      accq[j] = wmma_bf16(af, bq, accq[j]);
      v16bf bk = load_bt_frag(&sWk[j * 16][0], 32, 0, lane);
      acck[j] = wmma_bf16(af, bk, acck[j]);
      v16bf bv = load_bt_frag(&sWv[j * 16][0], 32, 0, lane);
      accv[j] = wmma_bf16(af, bv, accv[j]);
    }
    __syncthreads();
  }
  // C/D layout: VGPR r -> row (half*8+r), col coln
#pragma unroll
  for (int j = 0; j < 8; ++j)
#pragma unroll
    for (int r = 0; r < 8; ++r) {
      int m = m0 + wave * 16 + half * 8 + r;
      int n = j * 16 + coln;
      qo[(size_t)m * 128 + n] = (bf16_t)accq[j][r];
      ko[(size_t)m * 128 + n] = (bf16_t)acck[j][r];
      int b = m / T, t = m % T;  // T = 4096 (power of two)
      vo[(size_t)b * 128 * T + (size_t)n * T + t] = (bf16_t)accv[j][r];
    }
}

// ---------------- lambda scalar --------------------------------------------
__global__ void lam_kernel(const float* __restrict__ lq1,
                           const float* __restrict__ lq2,
                           const float* __restrict__ lk1,
                           const float* __restrict__ lk2,
                           float* __restrict__ lam_out, int d) {
  if (threadIdx.x == 0 && blockIdx.x == 0) {
    float s1 = 0.f, s2 = 0.f;
    for (int i = 0; i < d; ++i) { s1 += lq1[i] * lk1[i]; s2 += lq2[i] * lk2[i]; }
    float lam_init = 0.8f - 0.6f * __expf(-0.3f);  // DEPTH=2
    *lam_out = __expf(s1) - __expf(s2) + lam_init;
  }
}

// ---------------- fused differential flash attention -----------------------
// Grid: (T/128, B). 8 waves; wave owns 16 query rows. Key block Bc = 64.
__global__ __launch_bounds__(256) void diffattn_kernel(
    const bf16_t* __restrict__ qg, const bf16_t* __restrict__ kg,
    const bf16_t* __restrict__ vtg, const float* __restrict__ lam_p,
    float* __restrict__ out, int T) {
  __shared__ __align__(16) bf16_t sK[64][128];    // [s][d]  (B^T for QK^T)
  __shared__ __align__(16) bf16_t sVT[128][64];   // [h][s]  (B^T for PV)
  __shared__ __align__(16) bf16_t sP[8][16][64];  // per-wave prob staging

  const int tid = threadIdx.x, lane = tid & 31, wave = tid >> 5;
  const int half = lane >> 4, coln = lane & 15;
  const int q0 = blockIdx.x * 128;
  const size_t bo = (size_t)blockIdx.y * (size_t)T * 128;
  const bf16_t* q  = qg + bo;
  const bf16_t* k  = kg + bo;
  const bf16_t* vt = vtg + bo;  // [h][t] layout for this batch
  const float lam = *lam_p;

  v16bf qf1[2], qf2[2];
  {
    const bf16_t* qbase = q + (size_t)(q0 + wave * 16) * 128;
    qf1[0] = load_a_frag(qbase, 128, 0, lane);
    qf1[1] = load_a_frag(qbase, 128, 32, lane);
    qf2[0] = load_a_frag(qbase, 128, 64, lane);
    qf2[1] = load_a_frag(qbase, 128, 96, lane);
  }

  float m1[8], l1[8], m2[8], l2[8];
  v8f O1[8], O2[8];
#pragma unroll
  for (int r = 0; r < 8; ++r) { m1[r] = -1e30f; l1[r] = 0.f; m2[r] = -1e30f; l2[r] = 0.f; }
#pragma unroll
  for (int j = 0; j < 8; ++j)
#pragma unroll
    for (int r = 0; r < 8; ++r) { O1[j][r] = 0.f; O2[j][r] = 0.f; }

  const int nkb = (q0 + 128) / 64;  // causal: keys [0, q0+128)
  for (int kbi = 0; kbi < nkb; ++kbi) {
    const int s0 = kbi * 64;
    __syncthreads();  // everyone done reading previous tiles
    // K tile 64x128 bf16: pure copy -> async to LDS (1024 x 16B, 4/thread)
#pragma unroll
    for (int i = 0; i < 4; ++i) {
      int pos = tid + 256 * i;
      int s = pos >> 4, c = (pos & 15) << 3;
      async_copy_b128(k + (size_t)(s0 + s) * 128 + c, &sK[s][c]);
    }
    // V^T tile 128x64 bf16: pure copy from [h][t] layout
#pragma unroll
    for (int i = 0; i < 4; ++i) {
      int pos = tid + 256 * i;
      int h = pos >> 3, c = (pos & 7) << 3;
      async_copy_b128(vt + (size_t)h * T + s0 + c, &sVT[h][c]);
    }
    wait_async0();
    __syncthreads();

    // S1 = q1 k1^T, S2 = q2 k2^T   (16 x 64 per wave, K-dim = 64)
    v8f S1[4], S2[4];
#pragma unroll
    for (int j = 0; j < 4; ++j)
#pragma unroll
      for (int r = 0; r < 8; ++r) { S1[j][r] = 0.f; S2[j][r] = 0.f; }
#pragma unroll
    for (int j = 0; j < 4; ++j)
#pragma unroll
      for (int t = 0; t < 2; ++t) {
        v16bf b1 = load_bt_frag(&sK[j * 16][0], 128, t * 32, lane);
        S1[j] = wmma_bf16(qf1[t], b1, S1[j]);
        v16bf b2 = load_bt_frag(&sK[j * 16][0], 128, 64 + t * 32, lane);
        S2[j] = wmma_bf16(qf2[t], b2, S2[j]);
      }

    // scale + causal mask (element (row=half*8+r, key=s0+j*16+coln))
#pragma unroll
    for (int j = 0; j < 4; ++j)
#pragma unroll
      for (int r = 0; r < 8; ++r) {
        int qi = q0 + wave * 16 + half * 8 + r;
        int ki = s0 + j * 16 + coln;
        float x1 = S1[j][r] * 0.125f, x2 = S2[j][r] * 0.125f;
        if (ki > qi) { x1 = -1e30f; x2 = -1e30f; }
        S1[j][r] = x1; S2[j][r] = x2;
      }

    // two independent online softmaxes; row reduction over 16-lane half
    float a1[8], a2[8];
#pragma unroll
    for (int r = 0; r < 8; ++r) {
      float mx1 = S1[0][r], mx2 = S2[0][r];
#pragma unroll
      for (int j = 1; j < 4; ++j) { mx1 = fmaxf(mx1, S1[j][r]); mx2 = fmaxf(mx2, S2[j][r]); }
#pragma unroll
      for (int msk = 8; msk >= 1; msk >>= 1) {
        mx1 = fmaxf(mx1, __shfl_xor(mx1, msk, 32));
        mx2 = fmaxf(mx2, __shfl_xor(mx2, msk, 32));
      }
      float nm1 = fmaxf(m1[r], mx1), nm2 = fmaxf(m2[r], mx2);
      a1[r] = __expf(m1[r] - nm1); a2[r] = __expf(m2[r] - nm2);
      float sum1 = 0.f, sum2 = 0.f;
#pragma unroll
      for (int j = 0; j < 4; ++j) {
        float p1 = __expf(S1[j][r] - nm1), p2 = __expf(S2[j][r] - nm2);
        S1[j][r] = p1; S2[j][r] = p2; sum1 += p1; sum2 += p2;
      }
#pragma unroll
      for (int msk = 8; msk >= 1; msk >>= 1) {
        sum1 += __shfl_xor(sum1, msk, 32);
        sum2 += __shfl_xor(sum2, msk, 32);
      }
      l1[r] = l1[r] * a1[r] + sum1; l2[r] = l2[r] * a2[r] + sum2;
      m1[r] = nm1; m2[r] = nm2;
    }
#pragma unroll
    for (int j = 0; j < 8; ++j)
#pragma unroll
      for (int r = 0; r < 8; ++r) { O1[j][r] *= a1[r]; O2[j][r] *= a2[r]; }

    // PV half 1: D-layout -> A-layout via per-wave LDS staging
#pragma unroll
    for (int j = 0; j < 4; ++j)
#pragma unroll
      for (int r = 0; r < 8; ++r)
        sP[wave][half * 8 + r][j * 16 + coln] = (bf16_t)S1[j][r];
    {
      v16bf pa0 = load_a_frag(&sP[wave][0][0], 64, 0, lane);
      v16bf pa1 = load_a_frag(&sP[wave][0][0], 64, 32, lane);
#pragma unroll
      for (int j = 0; j < 8; ++j) {
        v16bf b0 = load_bt_frag(&sVT[j * 16][0], 64, 0, lane);
        O1[j] = wmma_bf16(pa0, b0, O1[j]);
        v16bf b1v = load_bt_frag(&sVT[j * 16][0], 64, 32, lane);
        O1[j] = wmma_bf16(pa1, b1v, O1[j]);
      }
    }
    // PV half 2 (wave-private sP reuse; DS ops are in-order per wave)
#pragma unroll
    for (int j = 0; j < 4; ++j)
#pragma unroll
      for (int r = 0; r < 8; ++r)
        sP[wave][half * 8 + r][j * 16 + coln] = (bf16_t)S2[j][r];
    {
      v16bf pa0 = load_a_frag(&sP[wave][0][0], 64, 0, lane);
      v16bf pa1 = load_a_frag(&sP[wave][0][0], 64, 32, lane);
#pragma unroll
      for (int j = 0; j < 8; ++j) {
        v16bf b0 = load_bt_frag(&sVT[j * 16][0], 64, 0, lane);
        O2[j] = wmma_bf16(pa0, b0, O2[j]);
        v16bf b1v = load_bt_frag(&sVT[j * 16][0], 64, 32, lane);
        O2[j] = wmma_bf16(pa1, b1v, O2[j]);
      }
    }
  }

  // out = O1/l1 - lam * O2/l2
#pragma unroll
  for (int j = 0; j < 8; ++j)
#pragma unroll
    for (int r = 0; r < 8; ++r) {
      int mrow = q0 + wave * 16 + half * 8 + r;
      int h = j * 16 + coln;
      float o = O1[j][r] / l1[r] - lam * (O2[j][r] / l2[r]);
      out[bo + (size_t)mrow * 128 + h] = o;
    }
}

// ---------------- host launch ----------------------------------------------
extern "C" void kernel_launch(void* const* d_in, const int* in_sizes, int n_in,
                              void* d_out, int out_size, void* d_ws, size_t ws_size,
                              hipStream_t stream) {
  (void)in_sizes; (void)n_in; (void)out_size; (void)ws_size;
  constexpr int Bb = 2, Tt = 4096, Cc = 2048, Hh = 128, Dd = 64;
  const float* x   = (const float*)d_in[0];
  const float* Wq  = (const float*)d_in[1];
  const float* Wk  = (const float*)d_in[2];
  const float* Wv  = (const float*)d_in[3];
  const float* lq1 = (const float*)d_in[4];
  const float* lq2 = (const float*)d_in[5];
  const float* lk1 = (const float*)d_in[6];
  const float* lk2 = (const float*)d_in[7];
  float* out = (float*)d_out;

  const size_t NTOK = (size_t)Bb * Tt;          // 8192
  bf16_t* qb = (bf16_t*)d_ws;                   // 2 MB  [tok][h]
  bf16_t* kb = qb + NTOK * Hh;                  // 2 MB  [tok][h]
  bf16_t* vb = kb + NTOK * Hh;                  // 2 MB  [b][h][t]
  float* lam = (float*)(vb + NTOK * Hh);

  dim3 bp(256), gp((unsigned)(NTOK / 128));
  qkv_proj_fused<<<gp, bp, 0, stream>>>(x, Wq, Wk, Wv, qb, kb, vb, Cc, Tt);
  lam_kernel<<<1, 32, 0, stream>>>(lq1, lq2, lk1, lk2, lam, Dd);
  dim3 ga(Tt / 128, Bb);
  diffattn_kernel<<<ga, bp, 0, stream>>>(qb, kb, vb, lam, out, Tt);
}
